// MLAttention_87574383165929
// MI455X (gfx1250) — matrix operand order repaired and checked by reading, not compiled, not run on hardware
//
#include <hip/hip_runtime.h>
#include <hip/hip_bf16.h>
#include <stdint.h>

// ---------------- Problem constants ----------------
#define SEQ        8192
#define D_MODEL    4096
#define Q_IN       1536
#define N_HEADS    64
#define HEAD_DIM   128
#define ROPE_DIM   64
#define TOP_K      2048
#define HAD_SCALE  0.08838834764831845f   // 1/sqrt(128)
#define SM_SCALE   0.08838834764831845f   // 1/sqrt(128)
#define HEAD_SCALE 0.125f                 // 1/sqrt(64)

typedef __attribute__((ext_vector_type(16))) __bf16 v16bf;
typedef __attribute__((ext_vector_type(8)))  float  v8f;
typedef int v4i_vec __attribute__((vector_size(16)));

union FragBF { uint32_t u[8]; v16bf v; };

// CDNA5 async global->LDS copy (ASYNCcnt-tracked), with fallback.
#if defined(__has_builtin)
#if __has_builtin(__builtin_amdgcn_global_load_async_to_lds_b128) && \
    __has_builtin(__builtin_amdgcn_s_wait_asynccnt)
#define HAVE_ASYNC_LDS 1
#endif
#endif

#ifdef HAVE_ASYNC_LDS
__device__ __forceinline__ void async_copy_b128(const void* g, void* l) {
  __builtin_amdgcn_global_load_async_to_lds_b128(
      (__attribute__((address_space(1))) v4i_vec*)g,
      (__attribute__((address_space(3))) v4i_vec*)l, 0, 0);
}
#endif

// A-matrix 16x32 bf16 fragment (ISA 7.12.2): lane m=lane&15, half=lane>>4
// elem K for vgpr v: (v>>2)*16 + half*8 + (v&3)*2   -> dwords: (v>>2)*8 + half*4 + (v&3)
__device__ __forceinline__ v16bf load_a_frag(const uint32_t* rowptr, int k0_dw, int half) {
  FragBF f;
#pragma unroll
  for (int v = 0; v < 8; ++v) f.u[v] = rowptr[k0_dw + (v >> 2) * 8 + half * 4 + (v & 3)];
  return f.v;
}

// B-matrix 32x16 bf16 fragment: lane n=lane&15, halfB=lane>>4
// elem K for vgpr v: (v>>2)*16 + halfB*8 + (v&3)*2  -> dwords: (v>>2)*8 + halfB*4 + (v&3)
__device__ __forceinline__ v16bf load_b_frag(const uint32_t* rowptr, int k0_dw, int half) {
  FragBF f;
#pragma unroll
  for (int v = 0; v < 8; ++v) f.u[v] = rowptr[k0_dw + (v >> 2) * 8 + half * 4 + (v & 3)];
  return f.v;
}

// ---------------- fp32 -> bf16 convert ----------------
__global__ void cvt_bf16_kernel(const float* __restrict__ in, __hip_bfloat16* __restrict__ out, int n) {
  int i = blockIdx.x * blockDim.x + threadIdx.x;
  int stride = gridDim.x * blockDim.x;
  for (; i < n; i += stride) out[i] = __float2bfloat16(in[i]);
}

// ---------------- k projection: kraw = x @ w_k^T  (SEQ x 128), WMMA bf16 ----------------
__global__ __launch_bounds__(256) void kproj_kernel(const uint32_t* __restrict__ xbf,
                                                    const uint32_t* __restrict__ wkbf,
                                                    float* __restrict__ kraw) {
  int q0   = blockIdx.x * 16;
  int wave = threadIdx.x >> 5;
  int lane = threadIdx.x & 31;
  int half = lane >> 4, mn = lane & 15;
  const uint32_t* arow = xbf  + (size_t)(q0 + mn) * (D_MODEL / 2);
  const uint32_t* brow = wkbf + (size_t)(wave * 16 + mn) * (D_MODEL / 2);
  v8f acc = {};
  for (int kt = 0; kt < D_MODEL / 32; ++kt) {
    v16bf a = load_a_frag(arow, kt * 16, half);
    v16bf b = load_b_frag(brow, kt * 16, half);
    acc = __builtin_amdgcn_wmma_f32_16x16x32_bf16(false, a, false, b, (short)0, acc, false, false);
  }
#pragma unroll
  for (int v = 0; v < 8; ++v) {
    int row = q0 + half * 8 + v;
    int col = wave * 16 + mn;
    kraw[(size_t)row * HEAD_DIM + col] = acc[v];
  }
}

// ---------------- k post: LayerNorm + RoPE + Hadamard -> bf16 ----------------
__global__ __launch_bounds__(128) void kpost_kernel(const float* __restrict__ kraw,
                                                    const float* __restrict__ gamma,
                                                    const float* __restrict__ beta,
                                                    const float* __restrict__ fcos,
                                                    const float* __restrict__ fsin,
                                                    __hip_bfloat16* __restrict__ kf) {
  __shared__ float arr[HEAD_DIM];
  __shared__ float stats[2];
  int r = blockIdx.x;
  int d = threadIdx.x;
  float v = kraw[(size_t)r * HEAD_DIM + d];
  arr[d] = v;
  __syncthreads();
  if (d == 0) { float s = 0.f; for (int i = 0; i < HEAD_DIM; ++i) s += arr[i]; stats[0] = s / HEAD_DIM; }
  __syncthreads();
  float mu = stats[0];
  float dv = v - mu;
  arr[d] = dv * dv;
  __syncthreads();
  if (d == 0) { float s = 0.f; for (int i = 0; i < HEAD_DIM; ++i) s += arr[i]; stats[1] = s / HEAD_DIM; }
  __syncthreads();
  float var = stats[1];
  float ln = dv * rsqrtf(var + 1e-5f) * gamma[d] + beta[d];
  __syncthreads();
  arr[d] = ln;
  __syncthreads();
  if (d < ROPE_DIM / 2) {
    int p = d;
    float x0 = arr[2 * p], x1 = arr[2 * p + 1];
    float c = fcos[(size_t)r * (ROPE_DIM / 2) + p];
    float s = fsin[(size_t)r * (ROPE_DIM / 2) + p];
    arr[2 * p]     = x0 * c - x1 * s;
    arr[2 * p + 1] = x0 * s + x1 * c;
  }
  __syncthreads();
  float s2 = 0.f;
  for (int j = 0; j < HEAD_DIM; ++j) {
    float a = arr[j];
    s2 += ((__popc(j & d) & 1) ? -a : a);   // Sylvester Hadamard sign
  }
  kf[(size_t)r * HEAD_DIM + d] = __float2bfloat16(s2 * HAD_SCALE);
}

// ---------------- w = x @ w_weights^T * HEAD_SCALE  (SEQ x 64) ----------------
__global__ __launch_bounds__(256) void wproj_kernel(const float* __restrict__ x,
                                                    const float* __restrict__ ww,
                                                    float* __restrict__ wout) {
  int gid  = blockIdx.x * 8 + (threadIdx.x >> 5);   // one wave per (t,h)
  int lane = threadIdx.x & 31;
  int t = gid >> 6, h = gid & 63;
  const float* xr = x  + (size_t)t * D_MODEL;
  const float* wr = ww + (size_t)h * D_MODEL;
  float s = 0.f;
  for (int i = lane; i < D_MODEL; i += 32) s += xr[i] * wr[i];
#pragma unroll
  for (int o = 16; o > 0; o >>= 1) s += __shfl_down(s, o);
  if (lane == 0) wout[(size_t)t * N_HEADS + h] = s * HEAD_SCALE;
}

// ---------------- fused q: per head WMMA proj -> weighted sum -> RoPE -> Hadamard ----------------
// q_eff[t] = HAD_SCALE * H * R_t * (sum_h w[t,h] * (q_input[t] @ w_q_h^T))   (linearity of RoPE/Hadamard)
// M-tile = 32 rows (two 16-row WMMA subtiles sharing each B fragment) to halve L2 traffic on w_q.
#define QM 32
__global__ __launch_bounds__(256) void qfused_kernel(const uint32_t* __restrict__ qinbf,
                                                     const uint32_t* __restrict__ wqbf,
                                                     const float* __restrict__ wbuf,
                                                     const float* __restrict__ fcos,
                                                     const float* __restrict__ fsin,
                                                     __hip_bfloat16* __restrict__ qeff) {
  // 96KB A tile + 8KB w tile; first 16KB of A-tile space reused for RoPE/Hadamard staging.
  __shared__ __align__(16) uint32_t smem[QM * (Q_IN / 2) + QM * N_HEADS];
  uint32_t* sA = smem;                                // QM x 768 dwords (bf16 pairs)
  float*    sW = (float*)(smem + QM * (Q_IN / 2));    // QM x 64
  float*    sQ = (float*)smem;                        // QM x 128 (reused after GEMM)

  int q0  = blockIdx.x * QM;
  int tid = threadIdx.x;

  const uint32_t* gA = qinbf + (size_t)q0 * (Q_IN / 2);   // contiguous QM*1536 bf16 region
#ifdef HAVE_ASYNC_LDS
  for (int c = tid; c < QM * (Q_IN / 2) / 4; c += 256)    // 16B chunks
    async_copy_b128((const char*)gA + (size_t)c * 16, (char*)sA + (size_t)c * 16);
#else
  for (int i = tid; i < QM * (Q_IN / 2); i += 256) sA[i] = gA[i];
#endif
  for (int i = tid; i < QM * N_HEADS; i += 256) {
    int r = i >> 6, h = i & 63;
    sW[i] = wbuf[(size_t)(q0 + r) * N_HEADS + h];
  }
#ifdef HAVE_ASYNC_LDS
  __builtin_amdgcn_s_wait_asynccnt(0);
#endif
  __syncthreads();

  int wave = tid >> 5, lane = tid & 31;
  int half = lane >> 4, mn = lane & 15;
  const uint32_t* arow0 = sA + (size_t)mn * (Q_IN / 2);
  const uint32_t* arow1 = sA + (size_t)(16 + mn) * (Q_IN / 2);
  float qacc0[8], qacc1[8];
#pragma unroll
  for (int v = 0; v < 8; ++v) { qacc0[v] = 0.f; qacc1[v] = 0.f; }

  for (int h = 0; h < N_HEADS; ++h) {
    const uint32_t* brow = wqbf + (size_t)(h * HEAD_DIM + wave * 16 + mn) * (Q_IN / 2);
    v8f acc0 = {}, acc1 = {};
    for (int kt = 0; kt < Q_IN / 32; ++kt) {
      v16bf b  = load_b_frag(brow, kt * 16, half);
      v16bf a0 = load_a_frag(arow0, kt * 16, half);
      acc0 = __builtin_amdgcn_wmma_f32_16x16x32_bf16(false, a0, false, b, (short)0, acc0, false, false);
      v16bf a1 = load_a_frag(arow1, kt * 16, half);
      acc1 = __builtin_amdgcn_wmma_f32_16x16x32_bf16(false, a1, false, b, (short)0, acc1, false, false);
    }
#pragma unroll
    for (int v = 0; v < 8; ++v) {
      qacc0[v] += sW[(half * 8 + v) * N_HEADS + h] * acc0[v];
      qacc1[v] += sW[(16 + half * 8 + v) * N_HEADS + h] * acc1[v];
    }
  }
  __syncthreads();   // all waves done reading sA before reuse as sQ
#pragma unroll
  for (int v = 0; v < 8; ++v) {
    sQ[(half * 8 + v) * HEAD_DIM + wave * 16 + mn]        = qacc0[v];
    sQ[(16 + half * 8 + v) * HEAD_DIM + wave * 16 + mn]   = qacc1[v];
  }
  __syncthreads();

  // RoPE on first 64 dims of each of QM rows
  for (int i = tid; i < QM * (ROPE_DIM / 2); i += 256) {
    int r = i >> 5, p = i & 31;
    float x0 = sQ[r * HEAD_DIM + 2 * p], x1 = sQ[r * HEAD_DIM + 2 * p + 1];
    float c = fcos[(size_t)(q0 + r) * (ROPE_DIM / 2) + p];
    float s = fsin[(size_t)(q0 + r) * (ROPE_DIM / 2) + p];
    sQ[r * HEAD_DIM + 2 * p]     = x0 * c - x1 * s;
    sQ[r * HEAD_DIM + 2 * p + 1] = x0 * s + x1 * c;
  }
  __syncthreads();

  // Hadamard + scale, write bf16
  for (int i = tid; i < QM * HEAD_DIM; i += 256) {
    int r = i >> 7, d = i & 127;
    float s = 0.f;
    for (int j = 0; j < HEAD_DIM; ++j) {
      float a = sQ[r * HEAD_DIM + j];
      s += ((__popc(j & d) & 1) ? -a : a);
    }
    qeff[(size_t)(q0 + r) * HEAD_DIM + d] = __float2bfloat16(s * HAD_SCALE);
  }
}

// ---------------- scores = q_eff @ k^T * SM_SCALE + causal mask ----------------
__global__ __launch_bounds__(256) void scores_kernel(const uint32_t* __restrict__ qeffbf,
                                                     const uint32_t* __restrict__ kfbf,
                                                     float* __restrict__ scores) {
  int q0   = blockIdx.x * 16;
  int wave = threadIdx.x >> 5;
  int lane = threadIdx.x & 31;
  int half = lane >> 4, mn = lane & 15;
  int k0 = (blockIdx.y * 8 + wave) * 16;
  const uint32_t* arow = qeffbf + (size_t)(q0 + mn) * (HEAD_DIM / 2);
  const uint32_t* brow = kfbf   + (size_t)(k0 + mn) * (HEAD_DIM / 2);
  v8f acc = {};
#pragma unroll
  for (int kt = 0; kt < HEAD_DIM / 32; ++kt) {
    v16bf a = load_a_frag(arow, kt * 16, half);
    v16bf b = load_b_frag(brow, kt * 16, half);
    acc = __builtin_amdgcn_wmma_f32_16x16x32_bf16(false, a, false, b, (short)0, acc, false, false);
  }
#pragma unroll
  for (int v = 0; v < 8; ++v) {
    int row = q0 + half * 8 + v;
    int col = k0 + mn;
    float val = acc[v] * SM_SCALE + ((col <= row) ? 0.f : -1.0e9f);
    scores[(size_t)row * SEQ + col] = val;
  }
}

// ---------------- per-row bitonic top-k (value desc, index asc ties) ----------------
__global__ __launch_bounds__(512) void topk_kernel(const float* __restrict__ scores,
                                                   int* __restrict__ out) {
  __shared__ unsigned long long keys[SEQ];   // 64 KB
  int row = blockIdx.x;
  const float* srow = scores + (size_t)row * SEQ;
  for (int i = threadIdx.x; i < SEQ; i += 512) {
    unsigned b   = __float_as_uint(srow[i]);
    unsigned ord = b ^ ((b >> 31) ? 0xFFFFFFFFu : 0x80000000u);  // ascending-float order
    unsigned dsc = ~ord;                                          // descending-float order
    keys[i] = ((unsigned long long)dsc << 32) | (unsigned)i;
  }
  __syncthreads();
  for (unsigned k = 2; k <= SEQ; k <<= 1) {
    for (unsigned j = k >> 1; j > 0; j >>= 1) {
      for (unsigned i = threadIdx.x; i < SEQ; i += 512) {
        unsigned ixj = i ^ j;
        if (ixj > i) {
          bool up = ((i & k) == 0);
          unsigned long long a = keys[i], c = keys[ixj];
          if ((a > c) == up) { keys[i] = c; keys[ixj] = a; }
        }
      }
      __syncthreads();
    }
  }
  for (int i = threadIdx.x; i < TOP_K; i += 512)
    out[(size_t)row * TOP_K + i] = (int)(keys[i] & 0xFFFFFFFFull);
}

// ---------------- host launcher ----------------
extern "C" void kernel_launch(void* const* d_in, const int* in_sizes, int n_in,
                              void* d_out, int out_size, void* d_ws, size_t ws_size,
                              hipStream_t stream) {
  (void)in_sizes; (void)n_in; (void)out_size; (void)ws_size;
  const float* x         = (const float*)d_in[0];   // 8192 x 4096
  const float* q_input   = (const float*)d_in[1];   // 8192 x 1536
  const float* w_q       = (const float*)d_in[2];   // 8192 x 1536
  const float* w_k       = (const float*)d_in[3];   // 128  x 4096
  const float* k_gamma   = (const float*)d_in[4];   // 128
  const float* k_beta    = (const float*)d_in[5];   // 128
  const float* w_weights = (const float*)d_in[6];   // 64 x 4096
  const float* fcos      = (const float*)d_in[7];   // 8192 x 32
  const float* fsin      = (const float*)d_in[8];   // 8192 x 32
  int* out = (int*)d_out;

  char* ws = (char*)d_ws;
  size_t off = 0;
  auto take = [&](size_t bytes) { char* p = ws + off; off += (bytes + 255) & ~(size_t)255; return p; };

  __hip_bfloat16* x_bf   = (__hip_bfloat16*)take((size_t)SEQ * D_MODEL * 2);
  __hip_bfloat16* qin_bf = (__hip_bfloat16*)take((size_t)SEQ * Q_IN * 2);
  __hip_bfloat16* wq_bf  = (__hip_bfloat16*)take((size_t)N_HEADS * HEAD_DIM * Q_IN * 2);
  __hip_bfloat16* wk_bf  = (__hip_bfloat16*)take((size_t)HEAD_DIM * D_MODEL * 2);
  float*          kraw   = (float*)take((size_t)SEQ * HEAD_DIM * 4);
  __hip_bfloat16* kf_bf  = (__hip_bfloat16*)take((size_t)SEQ * HEAD_DIM * 2);
  float*          wbuf   = (float*)take((size_t)SEQ * N_HEADS * 4);
  __hip_bfloat16* qef_bf = (__hip_bfloat16*)take((size_t)SEQ * HEAD_DIM * 2);
  float*          scores = (float*)take((size_t)SEQ * SEQ * 4);

  // 1) bf16 conversions
  cvt_bf16_kernel<<<4096, 256, 0, stream>>>(x, x_bf, SEQ * D_MODEL);
  cvt_bf16_kernel<<<4096, 256, 0, stream>>>(q_input, qin_bf, SEQ * Q_IN);
  cvt_bf16_kernel<<<4096, 256, 0, stream>>>(w_q, wq_bf, N_HEADS * HEAD_DIM * Q_IN);
  cvt_bf16_kernel<<<1024, 256, 0, stream>>>(w_k, wk_bf, HEAD_DIM * D_MODEL);

  // 2) k = x @ w_k^T   (WMMA)
  kproj_kernel<<<SEQ / 16, 256, 0, stream>>>((const uint32_t*)x_bf, (const uint32_t*)wk_bf, kraw);

  // 3) LN + RoPE + Hadamard on k
  kpost_kernel<<<SEQ, 128, 0, stream>>>(kraw, k_gamma, k_beta, fcos, fsin, kf_bf);

  // 4) w = x @ w_weights^T * HEAD_SCALE
  wproj_kernel<<<(SEQ * N_HEADS) / 8, 256, 0, stream>>>(x, w_weights, wbuf);

  // 5) fused q path (WMMA per head, weighted collapse, RoPE, Hadamard)
  qfused_kernel<<<SEQ / QM, 256, 0, stream>>>((const uint32_t*)qin_bf, (const uint32_t*)wq_bf,
                                              wbuf, fcos, fsin, qef_bf);

  // 6) scores = q_eff @ k^T * SM_SCALE + causal mask   (WMMA)
  scores_kernel<<<dim3(SEQ / 16, SEQ / 128), 256, 0, stream>>>((const uint32_t*)qef_bf,
                                                               (const uint32_t*)kf_bf, scores);

  // 7) per-row top-2048 indices (bitonic, jax tie order)
  topk_kernel<<<SEQ, 512, 0, stream>>>(scores, out);
}
